// ValkyrieAttention_41669772705994
// MI455X (gfx1250) — matrix operand order repaired and checked by reading, not compiled, not run on hardware
//
#include <hip/hip_runtime.h>
#include <hip/hip_bf16.h>
#include <math.h>
#include <type_traits>

// ---------------------------------------------------------------------------
// CDNA5 (gfx1250) fused attention block: bf16 WMMA + async LDS DMA.
// ---------------------------------------------------------------------------

typedef __attribute__((ext_vector_type(16))) __bf16 bf16x16;
typedef __attribute__((ext_vector_type(8)))  float  f32x8;

union Frag16 { bf16x16 v; uint4 q[2]; __bf16 h[16]; };

// A-fragment style: elements 0..7 at p[0..7], elements 8..15 at p[16..23]
__device__ __forceinline__ bf16x16 load_frag_gap(const __bf16* p) {
  Frag16 f;
  const uint4* u = reinterpret_cast<const uint4*>(p);
  f.q[0] = u[0];
  f.q[1] = u[2];
  return f.v;
}
// 16 contiguous elements (contiguous k order, used for both A and B)
__device__ __forceinline__ bf16x16 load_frag_contig(const __bf16* p) {
  Frag16 f;
  const uint4* u = reinterpret_cast<const uint4*>(p);
  f.q[0] = u[0];
  f.q[1] = u[1];
  return f.v;
}

__device__ __forceinline__ f32x8 wmma_bf16(bf16x16 a, bf16x16 b, f32x8 c) {
  return __builtin_amdgcn_wmma_f32_16x16x32_bf16(false, a, false, b,
                                                 (short)0, c, false, false);
}

// XOR-swizzle reductions across the 16-lane column groups (wave32).
__device__ __forceinline__ float redmax16(float x) {
  x = fmaxf(x, __int_as_float(__builtin_amdgcn_ds_swizzle(__float_as_int(x), 0x041f)));
  x = fmaxf(x, __int_as_float(__builtin_amdgcn_ds_swizzle(__float_as_int(x), 0x081f)));
  x = fmaxf(x, __int_as_float(__builtin_amdgcn_ds_swizzle(__float_as_int(x), 0x101f)));
  x = fmaxf(x, __int_as_float(__builtin_amdgcn_ds_swizzle(__float_as_int(x), 0x201f)));
  return x;
}
__device__ __forceinline__ float redsum16(float x) {
  x += __int_as_float(__builtin_amdgcn_ds_swizzle(__float_as_int(x), 0x041f));
  x += __int_as_float(__builtin_amdgcn_ds_swizzle(__float_as_int(x), 0x081f));
  x += __int_as_float(__builtin_amdgcn_ds_swizzle(__float_as_int(x), 0x101f));
  x += __int_as_float(__builtin_amdgcn_ds_swizzle(__float_as_int(x), 0x201f));
  return x;
}

// Async global->LDS DMA (CDNA5, ASYNCcnt-tracked). lds_addr is the group
// offset (low 32 bits of the generic shared address per the aperture rules).
__device__ __forceinline__ void async_copy_b128(unsigned lds_addr,
                                                const void* gaddr) {
  asm volatile("global_load_async_to_lds_b128 %0, %1, off"
               :: "v"(lds_addr), "v"(gaddr) : "memory");
}
__device__ __forceinline__ void wait_async_le12() {
  asm volatile("s_wait_asynccnt 0xc" ::: "memory");
}
__device__ __forceinline__ void wait_async_le0() {
  asm volatile("s_wait_asynccnt 0x0" ::: "memory");
}

// ---------------------------------------------------------------------------
// Kernel 0: zero the KV cache region of d_out (harness poisons with 0xAA).
// ---------------------------------------------------------------------------
__global__ void zero_kernel(float* __restrict__ p, size_t n) {
  size_t i = (size_t)blockIdx.x * blockDim.x + threadIdx.x;
  size_t stride = (size_t)gridDim.x * blockDim.x;
  for (; i < n; i += stride) p[i] = 0.0f;
}

// ---------------------------------------------------------------------------
// Kernel 1/5: C[M x N] (fp32, ldc) = A[M x K] @ W[K x N], bf16 WMMA math.
// Block tile 128x128, K-step 32, 256 threads = 8 waves (4x2), each wave
// owns a 32x64 quadrant (2x4 WMMA tiles). Register ping-pong double buffer:
// global loads for tile k+1 are issued before computing tile k.
// ---------------------------------------------------------------------------
template <typename AT>
__global__ __launch_bounds__(256) void gemm_kernel(
    const AT* __restrict__ A, const float* __restrict__ W,
    float* __restrict__ C, int M, int N, int K, int ldc) {
  using AVec = std::conditional_t<sizeof(AT) == 4, float4, uint2>;
  __shared__ __attribute__((aligned(16))) __bf16 As[2][128][32];  // [m][k]
  __shared__ __attribute__((aligned(16))) __bf16 Bs[2][128][32];  // [n][k]

  const int tid  = threadIdx.x;
  const int lane = tid & 31;
  const int wave = tid >> 5;
  const int bm = blockIdx.y * 128;
  const int bn = blockIdx.x * 128;
  const int wm = (wave >> 1) * 32;
  const int wn = (wave & 1) * 64;

  f32x8 acc[2][4] = {};
  AVec   areg[4];
  float4 breg[4];

  auto load_tiles = [&](int k0) {
#pragma unroll
    for (int i = 0; i < 4; ++i) {
      int idx = tid + (i << 8);                 // 1024 vec4 slots
      int r  = idx >> 3, c4 = (idx & 7) * 4;    // A: 128 rows x 8 vec4
      areg[i] = *reinterpret_cast<const AVec*>(&A[(size_t)(bm + r) * K + k0 + c4]);
      int kk = idx >> 5, n4 = (idx & 31) * 4;   // B: 32 rows x 32 vec4
      breg[i] = *reinterpret_cast<const float4*>(&W[(size_t)(k0 + kk) * N + bn + n4]);
    }
  };
  auto store_tiles = [&](int buf) {
#pragma unroll
    for (int i = 0; i < 4; ++i) {
      int idx = tid + (i << 8);
      int r  = idx >> 3, c4 = (idx & 7) * 4;
      union { __bf16 h[4]; uint2 u; } pk;
      if constexpr (sizeof(AT) == 4) {
        pk.h[0] = (__bf16)areg[i].x; pk.h[1] = (__bf16)areg[i].y;
        pk.h[2] = (__bf16)areg[i].z; pk.h[3] = (__bf16)areg[i].w;
      } else {
        pk.u = areg[i];
      }
      *reinterpret_cast<uint2*>(&As[buf][r][c4]) = pk.u;
      int kk = idx >> 5, n4 = (idx & 31) * 4;   // transpose W into [n][k]
      Bs[buf][n4 + 0][kk] = (__bf16)breg[i].x;
      Bs[buf][n4 + 1][kk] = (__bf16)breg[i].y;
      Bs[buf][n4 + 2][kk] = (__bf16)breg[i].z;
      Bs[buf][n4 + 3][kk] = (__bf16)breg[i].w;
    }
  };
  auto compute = [&](int buf) {
    bf16x16 af[2], bfv[4];
#pragma unroll
    for (int mt = 0; mt < 2; ++mt)
      af[mt] = load_frag_contig(&As[buf][wm + mt * 16 + (lane & 15)][(lane >> 4) * 16]);
#pragma unroll
    for (int nt = 0; nt < 4; ++nt)
      bfv[nt] = load_frag_contig(&Bs[buf][wn + nt * 16 + (lane & 15)][(lane >> 4) * 16]);
#pragma unroll
    for (int mt = 0; mt < 2; ++mt)
#pragma unroll
      for (int nt = 0; nt < 4; ++nt)
        acc[mt][nt] = wmma_bf16(af[mt], bfv[nt], acc[mt][nt]);
  };

  load_tiles(0);
  store_tiles(0);
  __syncthreads();
  int i = 0;
  for (int k0 = 0; k0 < K; k0 += 32, ++i) {
    const int buf = i & 1;
    const bool more = (k0 + 32) < K;
    if (more) load_tiles(k0 + 32);   // overlap next-tile global loads
    compute(buf);
    if (more) {
      __syncthreads();               // all waves finished reading buf^1 earlier
      store_tiles(buf ^ 1);
      __syncthreads();
    }
  }

  // epilogue (C layout: row = 8*(lane>>4)+r, col = lane&15)
#pragma unroll
  for (int mt = 0; mt < 2; ++mt)
#pragma unroll
    for (int nt = 0; nt < 4; ++nt)
#pragma unroll
      for (int r = 0; r < 8; ++r) {
        int row = bm + wm + mt * 16 + ((lane >> 4) << 3) + r;
        int col = bn + wn + nt * 16 + (lane & 15);
        C[(size_t)row * ldc + col] = acc[mt][nt][r];
      }
}

// ---------------------------------------------------------------------------
// Kernel 2: RoPE on q/k, scatter to bf16 layouts + fp32 KV caches.
//   qws  : bf16 [B,16,T,96]
//   kws  : bf16 [B,4,T,96]        (post-RoPE)
//   vwsT : bf16 [B,4,96,T]        (transposed for P@V B-fragments)
//   kcache/vcache : fp32 [B,4,4096,96]
// ---------------------------------------------------------------------------
__global__ __launch_bounds__(256) void rope_scatter_kernel(
    const float* __restrict__ qkv, const float* __restrict__ fcos,
    const float* __restrict__ fsin, const int* __restrict__ pos_ids,
    __bf16* __restrict__ qws, __bf16* __restrict__ kws,
    __bf16* __restrict__ vwsT, float* __restrict__ kcache,
    float* __restrict__ vcache) {
  const int bt = blockIdx.x;          // b*2048 + t
  const int b = bt >> 11;
  const int t = bt & 2047;
  const int pos = pos_ids[bt];
  const float* row = qkv + (size_t)bt * 2304;
  const float* cr = fcos + (size_t)pos * 48;
  const float* sr = fsin + (size_t)pos * 48;

  for (int idx = threadIdx.x; idx < 16 * 48; idx += 256) {
    int h = idx / 48, d2 = idx % 48;
    float c = cr[d2], s = sr[d2];
    float e = row[h * 96 + 2 * d2];
    float o = row[h * 96 + 2 * d2 + 1];
    float re = e * c - o * s;
    float ro = e * s + o * c;
    size_t base = (((size_t)(b * 16 + h) * 2048) + t) * 96 + 2 * d2;
    qws[base]     = (__bf16)re;
    qws[base + 1] = (__bf16)ro;
  }
  for (int idx = threadIdx.x; idx < 4 * 48; idx += 256) {
    int kh = idx / 48, d2 = idx % 48;
    float c = cr[d2], s = sr[d2];
    float e = row[1536 + kh * 96 + 2 * d2];
    float o = row[1536 + kh * 96 + 2 * d2 + 1];
    float re = e * c - o * s;
    float ro = e * s + o * c;
    size_t wb = (((size_t)(b * 4 + kh) * 2048) + t) * 96 + 2 * d2;
    kws[wb]     = (__bf16)re;
    kws[wb + 1] = (__bf16)ro;
    size_t cb = (((size_t)(b * 4 + kh) * 4096) + t) * 96 + 2 * d2;
    kcache[cb]     = re;
    kcache[cb + 1] = ro;
  }
  for (int idx = threadIdx.x; idx < 4 * 96; idx += 256) {
    int vh = idx / 96, d = idx % 96;
    float val = row[1920 + vh * 96 + d];
    vwsT[(((size_t)(b * 4 + vh) * 96) + d) * 2048 + t] = (__bf16)val;
    vcache[(((size_t)(b * 4 + vh) * 4096) + t) * 96 + d] = val;
  }
}

// ---------------------------------------------------------------------------
// Kernel 4: causal flash attention, bf16 WMMA, online softmax.
// grid = (T/64, B*16). 128 threads = 4 waves; each wave owns a 16-row stripe.
// K/V (and Q) tiles are staged with global_load_async_to_lds_b128 into
// double-buffered LDS; next tile's DMA overlaps current tile's WMMA+softmax.
// ---------------------------------------------------------------------------
__global__ __launch_bounds__(128) void attn_kernel(
    const __bf16* __restrict__ qws, const __bf16* __restrict__ kws,
    const __bf16* __restrict__ vwsT, __bf16* __restrict__ attnout) {
  __shared__ __attribute__((aligned(16))) __bf16 Qs[64][96];      // [t][d]
  __shared__ __attribute__((aligned(16))) __bf16 Ks[2][64][96];   // [t][d]
  __shared__ __attribute__((aligned(16))) __bf16 Vs[2][96][64];   // [d][t]
  __shared__ __attribute__((aligned(16))) __bf16 Ps[4][16][64];

  const int tid  = threadIdx.x;
  const int lane = tid & 31;
  const int wave = tid >> 5;
  const int qt0 = blockIdx.x * 64;
  const int bh  = blockIdx.y;        // b*16 + h
  const int b = bh >> 4, h = bh & 15;
  const int kv = h >> 2;             // GQA: 4 q heads per kv head
  const float scale = 0.10206207262f;  // 1/sqrt(96)

  const __bf16* qbase   = qws  + (((size_t)bh * 2048) + qt0) * 96;
  const __bf16* kbase_g = kws  + ((size_t)(b * 4 + kv) * 2048) * 96;
  const __bf16* vbase_g = vwsT + ((size_t)(b * 4 + kv) * 96) * 2048;

  auto issue_kv = [&](int kt, int buf) {
    const char* kg = reinterpret_cast<const char*>(kbase_g + (size_t)kt * 96);
    unsigned klds = (unsigned)(uintptr_t)&Ks[buf][0][0];
#pragma unroll
    for (int i = 0; i < 6; ++i) {            // 64*96 bf16 = 768 b128 chunks
      int idx = tid + (i << 7);
      async_copy_b128(klds + idx * 16, kg + idx * 16);
    }
    unsigned vlds = (unsigned)(uintptr_t)&Vs[buf][0][0];
#pragma unroll
    for (int i = 0; i < 6; ++i) {            // 96 rows x 8 chunks of 8 elems
      int idx = tid + (i << 7);
      int d = idx >> 3, t16 = (idx & 7) * 8;
      async_copy_b128(vlds + idx * 16,
                      vbase_g + (size_t)d * 2048 + kt + t16);
    }
  };

  // stage Q tile (async) + first K/V tile (async, buffer 0)
  {
    const char* qg = reinterpret_cast<const char*>(qbase);
    unsigned qlds = (unsigned)(uintptr_t)&Qs[0][0];
#pragma unroll
    for (int i = 0; i < 6; ++i) {
      int idx = tid + (i << 7);
      async_copy_b128(qlds + idx * 16, qg + idx * 16);
    }
  }
  issue_kv(0, 0);
  wait_async_le12();     // Q (issued first, in-order) is complete
  __syncthreads();

  bf16x16 qfrag[3];
#pragma unroll
  for (int c = 0; c < 3; ++c)
    qfrag[c] = load_frag_contig(&Qs[wave * 16 + (lane & 15)][c * 32 + (lane >> 4) * 16]);

  f32x8 o[6] = {};
  float mrow[8], lrow[8];
#pragma unroll
  for (int r = 0; r < 8; ++r) { mrow[r] = -INFINITY; lrow[r] = 0.0f; }
  const int qrow_base = qt0 + wave * 16 + ((lane >> 4) << 3);

  int it = 0;
  for (int kt0 = 0; kt0 < qt0 + 64; kt0 += 64, ++it) {
    const int buf = it & 1;
    const bool more = (kt0 + 64) <= qt0;
    if (more) issue_kv(kt0 + 64, buf ^ 1);   // buf^1 free since iter it-1 barrier
    if (more) wait_async_le12(); else wait_async_le0();  // current tile resident
    __syncthreads();

    // S = (Q @ K^T): 4 column tiles x 3 K-chunks
    f32x8 s[4] = {};
#pragma unroll
    for (int f = 0; f < 4; ++f) {
#pragma unroll
      for (int c = 0; c < 3; ++c) {
        bf16x16 kf = load_frag_contig(
            &Ks[buf][f * 16 + (lane & 15)][c * 32 + (lane >> 4) * 16]);
        s[f] = wmma_bf16(qfrag[c], kf, s[f]);
      }
    }
    // scale + causal mask
#pragma unroll
    for (int f = 0; f < 4; ++f) {
      int col = kt0 + f * 16 + (lane & 15);
#pragma unroll
      for (int r = 0; r < 8; ++r) {
        float v = s[f][r] * scale;
        s[f][r] = (col > qrow_base + r) ? -INFINITY : v;
      }
    }
    // online softmax (row stats across 16-lane column groups)
#pragma unroll
    for (int r = 0; r < 8; ++r) {
      float t0 = fmaxf(fmaxf(s[0][r], s[1][r]), fmaxf(s[2][r], s[3][r]));
      t0 = redmax16(t0);
      float mnew  = fmaxf(mrow[r], t0);
      float alpha = __expf(mrow[r] - mnew);
      mrow[r] = mnew;
      float p0 = __expf(s[0][r] - mnew); s[0][r] = p0;
      float p1 = __expf(s[1][r] - mnew); s[1][r] = p1;
      float p2 = __expf(s[2][r] - mnew); s[2][r] = p2;
      float p3 = __expf(s[3][r] - mnew); s[3][r] = p3;
      float rs = redsum16(p0 + p1 + p2 + p3);
      lrow[r] = lrow[r] * alpha + rs;
#pragma unroll
      for (int nf = 0; nf < 6; ++nf) o[nf][r] *= alpha;
    }
    // stash P (wave-private LDS; DS ops are in-order within a wave)
#pragma unroll
    for (int f = 0; f < 4; ++f)
#pragma unroll
      for (int r = 0; r < 8; ++r)
        Ps[wave][((lane >> 4) << 3) + r][f * 16 + (lane & 15)] = (__bf16)s[f][r];

    // O += P @ V : 6 d-tiles x 2 t-chunks
    bf16x16 pf[2];
#pragma unroll
    for (int c = 0; c < 2; ++c)
      pf[c] = load_frag_contig(&Ps[wave][lane & 15][c * 32 + (lane >> 4) * 16]);
#pragma unroll
    for (int nf = 0; nf < 6; ++nf) {
#pragma unroll
      for (int c = 0; c < 2; ++c) {
        bf16x16 vf = load_frag_contig(
            &Vs[buf][nf * 16 + (lane & 15)][c * 32 + (lane >> 4) * 16]);
        o[nf] = wmma_bf16(pf[c], vf, o[nf]);
      }
    }
    __syncthreads();   // all waves done with buf before it is refilled
  }

  // normalize and write [b, t, h*96 + d] for the Wo GEMM
#pragma unroll
  for (int nf = 0; nf < 6; ++nf)
#pragma unroll
    for (int r = 0; r < 8; ++r) {
      int row = qrow_base + r;
      float val = o[nf][r] / lrow[r];
      attnout[((size_t)(b * 2048 + row)) * 1536 + h * 96 + nf * 16 + (lane & 15)]
          = (__bf16)val;
    }
}

// ---------------------------------------------------------------------------
// Launch: zero caches -> QKV GEMMs -> RoPE/scatter -> attention -> Wo GEMM
// ---------------------------------------------------------------------------
extern "C" void kernel_launch(void* const* d_in, const int* in_sizes, int n_in,
                              void* d_out, int out_size, void* d_ws,
                              size_t ws_size, hipStream_t stream) {
  const float* x    = (const float*)d_in[0];
  const float* fcos = (const float*)d_in[1];
  const float* fsin = (const float*)d_in[2];
  const int*   pos  = (const int*)d_in[3];
  const float* Wq   = (const float*)d_in[4];
  const float* Wk   = (const float*)d_in[5];
  const float* Wv   = (const float*)d_in[6];
  const float* Wo   = (const float*)d_in[7];

  float* out    = (float*)d_out;                       // [2,2048,1536]
  float* kcache = out + (size_t)2 * 2048 * 1536;       // [2,4,4096,96]
  float* vcache = kcache + (size_t)2 * 4 * 4096 * 96;  // [2,4,4096,96]

  char* ws = (char*)d_ws;
  float*  qkv    = (float*)ws;                         // [4096, 2304] f32
  size_t  off    = (size_t)4096 * 2304 * 4;
  __bf16* qws    = (__bf16*)(ws + off); off += (size_t)2 * 16 * 2048 * 96 * 2;
  __bf16* kws    = (__bf16*)(ws + off); off += (size_t)2 * 4 * 2048 * 96 * 2;
  __bf16* vwsT   = (__bf16*)(ws + off); off += (size_t)2 * 4 * 96 * 2048 * 2;
  __bf16* attnws = (__bf16*)(ws + off);

  // 1) zero both caches (tail positions T..4095 must be exact zeros)
  zero_kernel<<<1024, 256, 0, stream>>>(kcache, (size_t)2 * 2 * 4 * 4096 * 96);

  // 2) QKV projections into one row-fused [4096 x 2304] buffer
  gemm_kernel<float><<<dim3(1536 / 128, 4096 / 128), 256, 0, stream>>>(
      x, Wq, qkv, 4096, 1536, 1536, 2304);
  gemm_kernel<float><<<dim3(384 / 128, 4096 / 128), 256, 0, stream>>>(
      x, Wk, qkv + 1536, 4096, 384, 1536, 2304);
  gemm_kernel<float><<<dim3(384 / 128, 4096 / 128), 256, 0, stream>>>(
      x, Wv, qkv + 1920, 4096, 384, 1536, 2304);

  // 3) RoPE + scatter (bf16 attention operands + fp32 KV caches)
  rope_scatter_kernel<<<4096, 256, 0, stream>>>(qkv, fcos, fsin, pos, qws, kws,
                                                vwsT, kcache, vcache);

  // 4) causal flash attention with WMMA + async LDS DMA
  attn_kernel<<<dim3(2048 / 64, 2 * 16), 128, 0, stream>>>(qws, kws, vwsT,
                                                           attnws);

  // 5) output projection -> d_out
  gemm_kernel<__bf16><<<dim3(1536 / 128, 4096 / 128), 256, 0, stream>>>(
      attnws, Wo, out, 4096, 1536, 1536, 1536);
}
